// HistoryEncoder_81544249082361
// MI455X (gfx1250) — compile-verified, matrix operand order checked by reading, not run on hardware
//
#include <hip/hip_runtime.h>
#include <hip/hip_bf16.h>

// ---------------------------------------------------------------------------
// MI455X (gfx1250) implementation. Compute-dominant convs + all dense linears
// run through v_wmma_f32_16x16x32_f16 (f16 in, f32 accumulate).
// LDS tiles are laid out so every WMMA fragment loads as 2x ds_load_b128;
// staging is two-phase (batched global loads -> batched cvt+LDS stores) with
// all n-side im2col address math hoisted out of the k-loop, and (where dims
// are exact multiples of the 64x64x64 tile) fully unguarded so EXEC stays
// all-ones.
// ---------------------------------------------------------------------------

typedef __attribute__((ext_vector_type(16))) _Float16 v16h;
typedef __attribute__((ext_vector_type(8)))  _Float16 v8h;
typedef __attribute__((ext_vector_type(4)))  _Float16 v4h;
typedef __attribute__((ext_vector_type(2)))  _Float16 v2h;
typedef __attribute__((ext_vector_type(8)))  float    v8f;
typedef __attribute__((ext_vector_type(4)))  float    v4f;

#define D_MODEL 512
#define D_STATE 64
#define D_CONV  4
#define D_INNER 1024
#define NHEADS  16
#define HEADDIM 64
#define NH_ATTN 8
#define STATE_DIM 14
#define BATCH 8
#define SEQ 64
#define BL 512           // BATCH*SEQ
#define CFEAT 512
#define ZW 2192          // 2*D_INNER + 2*D_STATE + NHEADS
#define XBCW 1152        // D_INNER + 2*D_STATE

#define LDP 72           // LDS row pitch in halfs (144B, 16B aligned, padded)

// ---------------- WMMA fragment helpers (16x16x32 f16) ---------------------
// Both A (M-major) and transposed-B (N-major) tiles use the same per-lane
// layout: lane l = row + 16*s; half i -> k = 8*s + i (i<8), 16+8*s+(i-8).
// Rows are LDP halfs apart -> each fragment = two 16B contiguous chunks.
__device__ __forceinline__ v16h frag_ld(const _Float16* base) {
  const int l = threadIdx.x & 31;
  const int row = l & 15;
  const int s = l >> 4;
  const _Float16* p = base + row * LDP + 8 * s;
  v8h lo = *(const v8h*)p;          // k = 8s .. 8s+7
  v8h hi = *(const v8h*)(p + 16);   // k = 16+8s .. 16+8s+7
  return __builtin_shufflevector(lo, hi, 0, 1, 2, 3, 4, 5, 6, 7,
                                 8, 9, 10, 11, 12, 13, 14, 15);
}
__device__ __forceinline__ v8f wmma_f16(v16h a, v16h b, v8f c) {
  return __builtin_amdgcn_wmma_f32_16x16x32_f16(false, a, false, b, (short)0, c,
                                                false, false);
}

// ---------------------------------------------------------------------------
// Implicit-GEMM 3x3 SAME conv on 7x7 maps.
//   M = Cout, K = Cin*9, N = Nimg*49.  A = weights (f32), B = im2col(input).
//   All of M, K, N are exact multiples of 64 for our shapes -> no guards.
//   Epilogue: +bias[row], ReLU, store f16 as out[img][co][49].
// ---------------------------------------------------------------------------
template <typename Tin>
__global__ __launch_bounds__(128) void conv3x3_wmma(
    const Tin* __restrict__ in, const float* __restrict__ w,
    const float* __restrict__ bias, _Float16* __restrict__ out,
    int Cin, int Cout, int Nimg) {
  __shared__ __align__(16) _Float16 As[64][LDP];   // [m][k]
  __shared__ __align__(16) _Float16 Bst[64][LDP];  // [n][k]  (transposed)
  const int K = Cin * 9;
  const int m0 = blockIdx.y * 64;
  const int n0 = blockIdx.x * 64;
  const int tid = threadIdx.x;
  const int wave = tid >> 5;
  const int wm = (wave >> 1) * 32;
  const int wn = (wave & 1) * 32;

  // ---- per-thread im2col n-side invariants (same for every k-tile) --------
  // B-staging slot t = tid + it*128:  n-index = t&63 = tid&63 (constant),
  // k-pair index kp = (tid>>6) + 2*it.
  const int nloc = tid & 63;
  const int n_g = n0 + nloc;
  const int img = n_g / 49, pix = n_g % 49;
  const int oh = pix / 7, ow = pix % 7;
  const Tin* __restrict__ inimg = in + (size_t)img * Cin * 49;
  const int kp0 = tid >> 6;

  // A-staging invariants: row = (tid>>4)+8*it, col4 = (tid&15)*4
  const int ar0 = tid >> 4;
  const int ac4 = (tid & 15) << 2;

  v8f acc00 = {}, acc01 = {}, acc10 = {}, acc11 = {};

  for (int kt = 0; kt < K; kt += 64) {
    // ---- phase 1: batch all global loads into registers
    v4f fa[8];
#pragma unroll
    for (int it = 0; it < 8; ++it) {
      const int r = ar0 + it * 8;
      fa[it] = *(const v4f*)(w + (size_t)(m0 + r) * K + kt + ac4);
    }
    float fb[32];
#pragma unroll
    for (int it = 0; it < 16; ++it) {
      const int k0 = kt + 2 * (kp0 + 2 * it);
      const int ci = k0 / 9, rc = k0 % 9;       // even-k decomposition
      const int ci1 = (rc == 8) ? ci + 1 : ci;  // odd k derived, no 2nd div
      const int rc1 = (rc == 8) ? 0 : rc + 1;
      const int cis[2] = {ci, ci1};
      const int rcs[2] = {rc, rc1};
#pragma unroll
      for (int u = 0; u < 2; ++u) {
        const int kr = rcs[u] / 3, kc = rcs[u] % 3;
        const int ih = oh + kr - 1, iw = ow + kc - 1;
        float v = 0.f;
        if (ih >= 0 && ih < 7 && iw >= 0 && iw < 7)
          v = (float)inimg[(size_t)cis[u] * 49 + ih * 7 + iw];
        fb[2 * it + u] = v;
      }
    }
    // ---- phase 2: convert + packed LDS stores
#pragma unroll
    for (int it = 0; it < 8; ++it) {
      v4h h;
      h[0] = (_Float16)fa[it][0]; h[1] = (_Float16)fa[it][1];
      h[2] = (_Float16)fa[it][2]; h[3] = (_Float16)fa[it][3];
      *(v4h*)&As[ar0 + it * 8][ac4] = h;
    }
#pragma unroll
    for (int it = 0; it < 16; ++it) {
      const int kp = kp0 + 2 * it;
      v2h h;
      h[0] = (_Float16)fb[2 * it];
      h[1] = (_Float16)fb[2 * it + 1];
      *(v2h*)&Bst[nloc][2 * kp] = h;
    }
    __syncthreads();
#pragma unroll
    for (int kk = 0; kk < 64; kk += 32) {
      v16h a0 = frag_ld(&As[wm][kk]);
      v16h a1 = frag_ld(&As[wm + 16][kk]);
      v16h b0 = frag_ld(&Bst[wn][kk]);
      v16h b1 = frag_ld(&Bst[wn + 16][kk]);
      acc00 = wmma_f16(a0, b0, acc00);
      acc01 = wmma_f16(a0, b1, acc01);
      acc10 = wmma_f16(a1, b0, acc10);
      acc11 = wmma_f16(a1, b1, acc11);
    }
    __syncthreads();
  }

  // epilogue: bias(row) + relu, scatter to [img][co][pix] as f16
  const int l = tid & 31;
  const int s = l >> 4;
  const v8f* accs[4] = {&acc00, &acc01, &acc10, &acc11};
#pragma unroll
  for (int si = 0; si < 2; ++si)
#pragma unroll
    for (int sj = 0; sj < 2; ++sj) {
      const v8f& a = *accs[si * 2 + sj];
      const int n = n0 + wn + sj * 16 + (l & 15);
      const int oimg = n / 49, opix = n % 49;
#pragma unroll
      for (int r = 0; r < 8; ++r) {
        const int m = m0 + wm + si * 16 + r + 8 * s;
        const float v = fmaxf(a[r] + bias[m], 0.f);
        out[((size_t)oimg * Cout + m) * 49 + opix] = (_Float16)v;
      }
    }
}

// ---------------------------------------------------------------------------
// Generic dense GEMM: C[M,N] = A[M,K] (f32, row-major) x B[K,N] (f32) +bias
// act: 0 = none, 1 = relu.  GUARD=false requires M,N,K multiples of 64.
// ---------------------------------------------------------------------------
template <bool GUARD>
__global__ __launch_bounds__(128) void gemm_wmma(
    const float* __restrict__ A, const float* __restrict__ Bw,
    const float* __restrict__ bias, float* __restrict__ C,
    int M, int N, int K, int act) {
  __shared__ __align__(16) _Float16 As[64][LDP];
  __shared__ __align__(16) _Float16 Bst[64][LDP];
  const int m0 = blockIdx.y * 64;
  const int n0 = blockIdx.x * 64;
  const int tid = threadIdx.x;
  const int wave = tid >> 5;
  const int wm = (wave >> 1) * 32;
  const int wn = (wave & 1) * 32;

  // staging invariants (unguarded path)
  const int nloc = tid & 63;
  const int kp0 = tid >> 6;
  const int ar0 = tid >> 4;
  const int ac4 = (tid & 15) << 2;
  const float* __restrict__ Bcol = Bw + n0 + nloc;  // column base, k strided

  v8f acc00 = {}, acc01 = {}, acc10 = {}, acc11 = {};

  for (int kt = 0; kt < K; kt += 64) {
    if (GUARD) {
      for (int t = tid; t < 64 * 64; t += 128) {
        const int r = t >> 6, c = t & 63;
        const int m = m0 + r, k = kt + c;
        As[r][c] = (_Float16)((m < M && k < K) ? A[(size_t)m * K + k] : 0.f);
      }
      for (int t = tid; t < 64 * 64; t += 128) {
        const int r = t >> 6, c = t & 63;   // r = k, c = n (coalesced on n)
        const int k = kt + r, n = n0 + c;
        Bst[c][r] = (_Float16)((k < K && n < N) ? Bw[(size_t)k * N + n] : 0.f);
      }
    } else {
      // ---- phase 1: batch all global loads into registers
      v4f fa[8];
#pragma unroll
      for (int it = 0; it < 8; ++it) {
        const int r = ar0 + it * 8;
        fa[it] = *(const v4f*)(A + (size_t)(m0 + r) * K + kt + ac4);
      }
      float fb[32];
#pragma unroll
      for (int it = 0; it < 16; ++it) {
        const int k = kt + 2 * (kp0 + 2 * it);
        fb[2 * it]     = Bcol[(size_t)k * N];
        fb[2 * it + 1] = Bcol[(size_t)(k + 1) * N];
      }
      // ---- phase 2: convert + packed LDS stores
#pragma unroll
      for (int it = 0; it < 8; ++it) {
        v4h h;
        h[0] = (_Float16)fa[it][0]; h[1] = (_Float16)fa[it][1];
        h[2] = (_Float16)fa[it][2]; h[3] = (_Float16)fa[it][3];
        *(v4h*)&As[ar0 + it * 8][ac4] = h;
      }
#pragma unroll
      for (int it = 0; it < 16; ++it) {
        const int kp = kp0 + 2 * it;
        v2h h;
        h[0] = (_Float16)fb[2 * it];
        h[1] = (_Float16)fb[2 * it + 1];
        *(v2h*)&Bst[nloc][2 * kp] = h;
      }
    }
    __syncthreads();
#pragma unroll
    for (int kk = 0; kk < 64; kk += 32) {
      v16h a0 = frag_ld(&As[wm][kk]);
      v16h a1 = frag_ld(&As[wm + 16][kk]);
      v16h b0 = frag_ld(&Bst[wn][kk]);
      v16h b1 = frag_ld(&Bst[wn + 16][kk]);
      acc00 = wmma_f16(a0, b0, acc00);
      acc01 = wmma_f16(a0, b1, acc01);
      acc10 = wmma_f16(a1, b0, acc10);
      acc11 = wmma_f16(a1, b1, acc11);
    }
    __syncthreads();
  }

  const int l = tid & 31;
  const int s = l >> 4;
  const v8f* accs[4] = {&acc00, &acc01, &acc10, &acc11};
#pragma unroll
  for (int si = 0; si < 2; ++si)
#pragma unroll
    for (int sj = 0; sj < 2; ++sj) {
      const v8f& a = *accs[si * 2 + sj];
      const int n = n0 + wn + sj * 16 + (l & 15);
#pragma unroll
      for (int r = 0; r < 8; ++r) {
        const int m = m0 + wm + si * 16 + r + 8 * s;
        if (!GUARD || (m < M && n < N)) {
          float v = a[r];
          if (bias) v += bias[n];
          if (act == 1) v = fmaxf(v, 0.f);
          C[(size_t)m * N + n] = v;
        }
      }
    }
}

// ---------------------------------------------------------------------------
// Small VALU kernels
// ---------------------------------------------------------------------------
__global__ void pool49_kernel(const _Float16* __restrict__ in,
                              float* __restrict__ out, int total) {
  const int i = blockIdx.x * blockDim.x + threadIdx.x;
  if (i >= total) return;
  const _Float16* p = in + (size_t)i * 49;
  float s = 0.f;
#pragma unroll
  for (int j = 0; j < 49; ++j) s += (float)p[j];
  out[i] = s * (1.f / 49.f);
}

// LayerNorm(512) + ReLU, written with column offset/stride (camera interleave)
__global__ __launch_bounds__(128) void ln_relu_kernel(
    const float* __restrict__ in, const float* __restrict__ g,
    const float* __restrict__ b, float* __restrict__ out,
    int D, int outStride, int outOffset) {
  __shared__ float red[128];
  const int row = blockIdx.x;
  const float* x = in + (size_t)row * D;
  float s = 0.f;
  for (int d = threadIdx.x; d < D; d += 128) s += x[d];
  red[threadIdx.x] = s;
  __syncthreads();
  for (int off = 64; off > 0; off >>= 1) {
    if (threadIdx.x < off) red[threadIdx.x] += red[threadIdx.x + off];
    __syncthreads();
  }
  const float mean = red[0] / D;
  __syncthreads();
  float v = 0.f;
  for (int d = threadIdx.x; d < D; d += 128) {
    const float t = x[d] - mean;
    v += t * t;
  }
  red[threadIdx.x] = v;
  __syncthreads();
  for (int off = 64; off > 0; off >>= 1) {
    if (threadIdx.x < off) red[threadIdx.x] += red[threadIdx.x + off];
    __syncthreads();
  }
  const float rstd = rsqrtf(red[0] / D + 1e-5f);
  float* o = out + (size_t)row * outStride + outOffset;
  for (int d = threadIdx.x; d < D; d += 128)
    o[d] = fmaxf((x[d] - mean) * rstd * g[d] + b[d], 0.f);
}

// 2-token camera self-attention; grid (BL, NH_ATTN), block 64
__global__ __launch_bounds__(64) void cca_attn_kernel(
    const float* __restrict__ qh, const float* __restrict__ kh,
    const float* __restrict__ vh, float* __restrict__ o) {
  const int bl = blockIdx.x, h = blockIdx.y, d = threadIdx.x;
  __shared__ float sdot[4][64];
  __shared__ float sc[4];
  const size_t r0 = ((size_t)bl * 2) * D_MODEL + h * 64;
  const size_t r1 = ((size_t)bl * 2 + 1) * D_MODEL + h * 64;
  const float q0d = qh[r0 + d], q1d = qh[r1 + d];
  const float k0d = kh[r0 + d], k1d = kh[r1 + d];
  sdot[0][d] = q0d * k0d; sdot[1][d] = q0d * k1d;
  sdot[2][d] = q1d * k0d; sdot[3][d] = q1d * k1d;
  __syncthreads();
  if (d < 4) {
    float t = 0.f;
    for (int j = 0; j < 64; ++j) t += sdot[d][j];
    sc[d] = t * 0.125f;  // 1/sqrt(64)
  }
  __syncthreads();
  float m0 = fmaxf(sc[0], sc[1]);
  float e0 = expf(sc[0] - m0), e1 = expf(sc[1] - m0);
  float i0 = 1.f / (e0 + e1);
  float m1 = fmaxf(sc[2], sc[3]);
  float e2 = expf(sc[2] - m1), e3 = expf(sc[3] - m1);
  float i1 = 1.f / (e2 + e3);
  const float v0d = vh[r0 + d], v1d = vh[r1 + d];
  o[r0 + d] = e0 * i0 * v0d + e1 * i0 * v1d;
  o[r1 + d] = e2 * i1 * v0d + e3 * i1 * v1d;
}

__global__ void mean_pairs_kernel(const float* __restrict__ oc,
                                  float* __restrict__ x) {
  const int i = blockIdx.x * blockDim.x + threadIdx.x;
  if (i >= BL * D_MODEL) return;
  const int row = i >> 9, d = i & 511;
  x[i] = 0.5f * (oc[((size_t)row * 2) * D_MODEL + d] +
                 oc[((size_t)row * 2 + 1) * D_MODEL + d]);
}

__global__ void kvproj_kernel(const float* __restrict__ obs,
                              const float* __restrict__ W,
                              const float* __restrict__ b,
                              float* __restrict__ kv) {
  const int i = blockIdx.x * blockDim.x + threadIdx.x;
  if (i >= BL * D_MODEL) return;
  const int t = i >> 9, n = i & 511;
  float s = b[n];
#pragma unroll
  for (int j = 0; j < STATE_DIM; ++j)
    s += obs[t * STATE_DIM + j] * W[j * D_MODEL + n];
  kv[i] = s;
}

// 64-key cross-attention; grid (SEQ, BATCH, NH_ATTN), block 64
__global__ __launch_bounds__(64) void cma_attn_kernel(
    const float* __restrict__ qh, const float* __restrict__ kh,
    const float* __restrict__ vh, float* __restrict__ o) {
  const int qi = blockIdx.x, b = blockIdx.y, h = blockIdx.z, t = threadIdx.x;
  __shared__ float qv[64], sc[64];
  __shared__ float sinv;
  const size_t base = (size_t)b * SEQ * D_MODEL;
  qv[t] = qh[base + (size_t)qi * D_MODEL + h * 64 + t];
  __syncthreads();
  float s = 0.f;
  const float* krow = kh + base + (size_t)t * D_MODEL + h * 64;
#pragma unroll 8
  for (int d = 0; d < 64; ++d) s += qv[d] * krow[d];
  sc[t] = s * 0.125f;
  __syncthreads();
  if (t == 0) {
    float mx = -1e30f;
    for (int j = 0; j < 64; ++j) mx = fmaxf(mx, sc[j]);
    float sm = 0.f;
    for (int j = 0; j < 64; ++j) { sc[j] = expf(sc[j] - mx); sm += sc[j]; }
    sinv = 1.f / sm;
  }
  __syncthreads();
  float acc = 0.f;
#pragma unroll 8
  for (int j = 0; j < 64; ++j)
    acc += sc[j] * vh[base + (size_t)j * D_MODEL + h * 64 + t];
  o[base + (size_t)qi * D_MODEL + h * 64 + t] = acc * sinv;
}

__global__ void dtda_kernel(const float* __restrict__ zx,
                            const float* __restrict__ dt_bias,
                            const float* __restrict__ A_log,
                            float* __restrict__ dt_o, float* __restrict__ dA_o) {
  const int i = blockIdx.x * blockDim.x + threadIdx.x;
  if (i >= BL * NHEADS) return;
  const int bl = i >> 4, h = i & 15;
  const float x = zx[(size_t)bl * ZW + (ZW - NHEADS) + h] + dt_bias[h];
  const float dt = (x > 20.f) ? x : log1pf(expf(x));
  dt_o[i] = dt;
  dA_o[i] = expf(dt * (-expf(A_log[h])));
}

__global__ void convsilu_kernel(const float* __restrict__ zx,
                                const float* __restrict__ w,
                                const float* __restrict__ b,
                                float* __restrict__ out) {
  const int i = blockIdx.x * blockDim.x + threadIdx.x;
  if (i >= BL * XBCW) return;
  const int bl = i / XBCW, ch = i % XBCW;
  const int bb = bl >> 6, l = bl & 63;
  float s = b[ch];
#pragma unroll
  for (int k = 0; k < D_CONV; ++k) {
    const int t = l - (D_CONV - 1) + k;
    if (t >= 0)
      s += zx[((size_t)(bb * SEQ + t)) * ZW + D_INNER + ch] * w[ch * D_CONV + k];
  }
  out[i] = s / (1.f + expf(-s));  // silu
}

// SSM scan: grid (NHEADS, BATCH), block HEADDIM; thread p holds h[p, 0..63]
__global__ __launch_bounds__(64) void ssm_scan_kernel(
    const float* __restrict__ conv, const float* __restrict__ dt,
    const float* __restrict__ dA, const float* __restrict__ Dp,
    float* __restrict__ y) {
  const int h = blockIdx.x, b = blockIdx.y, p = threadIdx.x;
  float hs[D_STATE];
#pragma unroll
  for (int n = 0; n < D_STATE; ++n) hs[n] = 0.f;
  __shared__ float Bs[D_STATE], Cs[D_STATE];
  const float Dh = Dp[h];
  for (int t = 0; t < SEQ; ++t) {
    const int bl = b * SEQ + t;
    Bs[p] = conv[(size_t)bl * XBCW + D_INNER + p];
    Cs[p] = conv[(size_t)bl * XBCW + D_INNER + D_STATE + p];
    __syncthreads();
    const float xv = conv[(size_t)bl * XBCW + h * HEADDIM + p];
    const float dbx = dt[bl * NHEADS + h] * xv;
    const float dAv = dA[bl * NHEADS + h];
    float acc = 0.f;
#pragma unroll
    for (int n = 0; n < D_STATE; ++n) {
      hs[n] = hs[n] * dAv + dbx * Bs[n];
      acc += hs[n] * Cs[n];
    }
    y[(size_t)bl * D_INNER + h * HEADDIM + p] = acc + Dh * xv;
    __syncthreads();
  }
}

// y * silu(z), RMSNorm over 1024, * norm_w
__global__ __launch_bounds__(256) void gate_rms_kernel(
    const float* __restrict__ y, const float* __restrict__ zx,
    const float* __restrict__ nw, float* __restrict__ out) {
  __shared__ float red[256];
  const int bl = blockIdx.x;
  float vals[4];
  float ss = 0.f;
#pragma unroll
  for (int q = 0; q < 4; ++q) {
    const int d = threadIdx.x + q * 256;
    const float z = zx[(size_t)bl * ZW + d];
    const float v = y[(size_t)bl * D_INNER + d] * (z / (1.f + expf(-z)));
    vals[q] = v;
    ss += v * v;
  }
  red[threadIdx.x] = ss;
  __syncthreads();
  for (int off = 128; off > 0; off >>= 1) {
    if (threadIdx.x < off) red[threadIdx.x] += red[threadIdx.x + off];
    __syncthreads();
  }
  const float r = rsqrtf(red[0] / D_INNER + 1e-5f);
#pragma unroll
  for (int q = 0; q < 4; ++q) {
    const int d = threadIdx.x + q * 256;
    out[(size_t)bl * D_INNER + d] = vals[q] * r * nw[d];
  }
}

// ---------------------------------------------------------------------------
// Host orchestration
// ---------------------------------------------------------------------------
static inline dim3 gemm_grid(int M, int N) {
  return dim3((N + 63) / 64, (M + 63) / 64);
}

extern "C" void kernel_launch(void* const* d_in, const int* in_sizes, int n_in,
                              void* d_out, int out_size, void* d_ws, size_t ws_size,
                              hipStream_t stream) {
  (void)in_sizes; (void)n_in; (void)out_size; (void)ws_size;
  // inputs (setup_inputs order)
  const float* obs      = (const float*)d_in[0];
  const float* img0     = (const float*)d_in[1];
  const float* img1     = (const float*)d_in[2];
  const float* c1w      = (const float*)d_in[3];
  const float* c1b      = (const float*)d_in[4];
  const float* c2w      = (const float*)d_in[5];
  const float* c2b      = (const float*)d_in[6];
  const float* linw     = (const float*)d_in[7];
  const float* linb     = (const float*)d_in[8];
  const float* lng      = (const float*)d_in[9];
  const float* lnb      = (const float*)d_in[10];
  const float* cca_wq   = (const float*)d_in[11]; const float* cca_bq = (const float*)d_in[12];
  const float* cca_wk   = (const float*)d_in[13]; const float* cca_bk = (const float*)d_in[14];
  const float* cca_wv   = (const float*)d_in[15]; const float* cca_bv = (const float*)d_in[16];
  const float* cca_wo   = (const float*)d_in[17]; const float* cca_bo = (const float*)d_in[18];
  const float* cma_wq   = (const float*)d_in[19]; const float* cma_bq = (const float*)d_in[20];
  const float* cma_wk   = (const float*)d_in[21]; const float* cma_bk = (const float*)d_in[22];
  const float* cma_wv   = (const float*)d_in[23]; const float* cma_bv = (const float*)d_in[24];
  const float* cma_wo   = (const float*)d_in[25]; const float* cma_bo = (const float*)d_in[26];
  const float* proj_w   = (const float*)d_in[27]; const float* proj_b = (const float*)d_in[28];
  const float* low_w    = (const float*)d_in[29]; const float* low_b  = (const float*)d_in[30];
  const float* m_in_w   = (const float*)d_in[31];
  const float* m_conv_w = (const float*)d_in[32];
  const float* m_conv_b = (const float*)d_in[33];
  const float* m_dtb    = (const float*)d_in[34];
  const float* m_Alog   = (const float*)d_in[35];
  const float* m_D      = (const float*)d_in[36];
  const float* m_nw     = (const float*)d_in[37];
  const float* m_out_w  = (const float*)d_in[38];

  // workspace bump allocator (256B aligned)
  char* base = (char*)d_ws;
  size_t off = 0;
  auto alloc = [&](size_t bytes) -> void* {
    void* p = base + off;
    off = (off + bytes + 255) & ~(size_t)255;
    return p;
  };
  _Float16* conv1o = (_Float16*)alloc((size_t)BL * 512 * 49 * 2);
  _Float16* conv2o = (_Float16*)alloc((size_t)BL * 256 * 49 * 2);
  float* pooled = (float*)alloc((size_t)BL * 256 * 4);
  float* linout = (float*)alloc((size_t)BL * 512 * 4);
  float* feats  = (float*)alloc((size_t)BL * 2 * 512 * 4);   // (bl, cam, d)
  float* qh  = (float*)alloc((size_t)BL * 2 * 512 * 4);
  float* kh  = (float*)alloc((size_t)BL * 2 * 512 * 4);
  float* vh  = (float*)alloc((size_t)BL * 2 * 512 * 4);
  float* oat = (float*)alloc((size_t)BL * 2 * 512 * 4);
  float* oc  = (float*)alloc((size_t)BL * 2 * 512 * 4);
  float* x   = (float*)alloc((size_t)BL * 512 * 4);
  float* qx  = (float*)alloc((size_t)BL * 512 * 4);
  float* kv  = (float*)alloc((size_t)BL * 512 * 4);
  float* qh2 = (float*)alloc((size_t)BL * 512 * 4);
  float* kh2 = (float*)alloc((size_t)BL * 512 * 4);
  float* vh2 = (float*)alloc((size_t)BL * 512 * 4);
  float* o2  = (float*)alloc((size_t)BL * 512 * 4);
  float* xseq= (float*)alloc((size_t)BL * 512 * 4);
  float* zx  = (float*)alloc((size_t)BL * ZW * 4);
  float* dtb = (float*)alloc((size_t)BL * NHEADS * 4);
  float* dab = (float*)alloc((size_t)BL * NHEADS * 4);
  float* cvo = (float*)alloc((size_t)BL * XBCW * 4);
  float* ybuf= (float*)alloc((size_t)BL * D_INNER * 4);
  float* gbuf= (float*)alloc((size_t)BL * D_INNER * 4);

  const dim3 blk128(128), blk64(64), blk256(256);
  const int Nc = BL * 49;                 // 25088 = 392*64 GEMM columns per conv
  const dim3 cgrid((Nc + 63) / 64, 512 / 64);
  const dim3 cgrid2((Nc + 63) / 64, 256 / 64);

  // ---------------- spatial adapter, per camera ----------------
  const float* imgs[2] = {img0, img1};
  for (int cam = 0; cam < 2; ++cam) {
    conv3x3_wmma<float><<<cgrid, blk128, 0, stream>>>(imgs[cam], c1w, c1b,
                                                      conv1o, CFEAT, 512, BL);
    conv3x3_wmma<_Float16><<<cgrid2, blk128, 0, stream>>>(conv1o, c2w, c2b,
                                                          conv2o, 512, 256, BL);
    pool49_kernel<<<(BL * 256 + 255) / 256, blk256, 0, stream>>>(conv2o, pooled,
                                                                 BL * 256);
    gemm_wmma<false><<<gemm_grid(BL, 512), blk128, 0, stream>>>(pooled, linw, linb,
                                                                linout, BL, 512, 256, 0);
    ln_relu_kernel<<<BL, blk128, 0, stream>>>(linout, lng, lnb, feats, 512,
                                              1024, cam * 512);
  }

  // ---------------- camera cross-attention (2 tokens) ----------------
  gemm_wmma<false><<<gemm_grid(2 * BL, 512), blk128, 0, stream>>>(feats, cca_wq, cca_bq, qh, 2 * BL, 512, 512, 0);
  gemm_wmma<false><<<gemm_grid(2 * BL, 512), blk128, 0, stream>>>(feats, cca_wk, cca_bk, kh, 2 * BL, 512, 512, 0);
  gemm_wmma<false><<<gemm_grid(2 * BL, 512), blk128, 0, stream>>>(feats, cca_wv, cca_bv, vh, 2 * BL, 512, 512, 0);
  cca_attn_kernel<<<dim3(BL, NH_ATTN), blk64, 0, stream>>>(qh, kh, vh, oat);
  gemm_wmma<false><<<gemm_grid(2 * BL, 512), blk128, 0, stream>>>(oat, cca_wo, cca_bo, oc, 2 * BL, 512, 512, 0);
  mean_pairs_kernel<<<(BL * 512 + 255) / 256, blk256, 0, stream>>>(oc, x);

  // ---------------- proj + cross-modal attention ----------------
  gemm_wmma<false><<<gemm_grid(BL, 512), blk128, 0, stream>>>(x, proj_w, proj_b, qx, BL, 512, 512, 0);
  kvproj_kernel<<<(BL * 512 + 255) / 256, blk256, 0, stream>>>(obs, low_w, low_b, kv);
  gemm_wmma<false><<<gemm_grid(BL, 512), blk128, 0, stream>>>(qx, cma_wq, cma_bq, qh2, BL, 512, 512, 0);
  gemm_wmma<false><<<gemm_grid(BL, 512), blk128, 0, stream>>>(kv, cma_wk, cma_bk, kh2, BL, 512, 512, 0);
  gemm_wmma<false><<<gemm_grid(BL, 512), blk128, 0, stream>>>(kv, cma_wv, cma_bv, vh2, BL, 512, 512, 0);
  cma_attn_kernel<<<dim3(SEQ, BATCH, NH_ATTN), blk64, 0, stream>>>(qh2, kh2, vh2, o2);
  gemm_wmma<false><<<gemm_grid(BL, 512), blk128, 0, stream>>>(o2, cma_wo, cma_bo, xseq, BL, 512, 512, 0);

  // ---------------- Mamba2 ----------------
  gemm_wmma<true><<<gemm_grid(BL, ZW), blk128, 0, stream>>>(xseq, m_in_w, nullptr, zx, BL, ZW, 512, 0);
  dtda_kernel<<<(BL * NHEADS + 255) / 256, blk256, 0, stream>>>(zx, m_dtb, m_Alog, dtb, dab);
  convsilu_kernel<<<(BL * XBCW + 255) / 256, blk256, 0, stream>>>(zx, m_conv_w, m_conv_b, cvo);
  ssm_scan_kernel<<<dim3(NHEADS, BATCH), blk64, 0, stream>>>(cvo, dtb, dab, m_D, ybuf);
  gate_rms_kernel<<<BL, blk256, 0, stream>>>(ybuf, zx, m_nw, gbuf);
  gemm_wmma<false><<<gemm_grid(BL, 512), blk128, 0, stream>>>(gbuf, m_out_w, nullptr,
                                                              (float*)d_out, BL, 512, D_INNER, 0);
}
